// VectorQuantizerEMA_6597069767086
// MI455X (gfx1250) — matrix operand order repaired and checked by reading, not compiled, not run on hardware
//
#include <hip/hip_runtime.h>
#include <math.h>

typedef __attribute__((ext_vector_type(16))) _Float16 v16h;
typedef __attribute__((ext_vector_type(8)))  float    v8f;

#define N_TOT   16384
#define K_CODES 4096
#define DIM     256
#define ZPAD    260   // LDS row stride pad: rows land 4 banks apart -> conflict-free

// ---- d_out flat offsets (floats), tuple order: loss, z_q_out, perplexity, encodings, indices
#define OUT_LOSS 0
#define OUT_ZQ   1
#define OUT_PERP 4194305ull
#define OUT_ENC  4194306ull
#define OUT_IDX  71303170ull

// ---- workspace byte offsets
#define WS_CB    0                      // 4096*256 f16 = 2 MB packed codebook
#define WS_IDX   (2u*1024u*1024u)       // 16384 int
#define WS_HIST  (WS_IDX + 65536u)      // 4096 int
#define WS_LOSS  (WS_HIST + 16384u)     // 512 float (per-block loss partials)

__global__ __launch_bounds__(256) void init_kernel(int* __restrict__ hist) {
    for (int k = threadIdx.x; k < K_CODES; k += 256) hist[k] = 0;
}

// Normalize codebook rows and pack f16 into a WMMA B-operand layout with contiguous k order:
// tile t = code/16, column j = code%16; per 32-K chunk c:
//   lane l = j + (kk&16), element p = kk&15  (lanes 0-15: K 0..15, lanes 16-31: K 16..31)
__global__ __launch_bounds__(256) void cb_norm_kernel(const float* __restrict__ embed,
                                                      _Float16* __restrict__ cb) {
    __shared__ float red[256];
    const int n = blockIdx.x;       // code row
    const int k = threadIdx.x;      // dim
    const float x = embed[(size_t)n * DIM + k];
    red[k] = x * x;
    __syncthreads();
    for (int s = 128; s > 0; s >>= 1) { if (k < s) red[k] += red[k + s]; __syncthreads(); }
    const float scale = 1.0f / fmaxf(sqrtf(red[0]), 1e-12f);
    const int t = n >> 4, j = n & 15;
    const int c = k >> 5, kk = k & 31;
    const int l = j + (kk & 16);
    const int p = kk & 15;
    cb[((size_t)((t * 8 + c) * 32 + l)) * 16 + p] = (_Float16)(x * scale);
}

// One block = 32 z-rows (two 16-row A tiles per wave); 8 waves split the 256 column tiles.
// Two independent accumulator chains per B load: 2x WMMA density per byte of L2 traffic,
// 2x ILP in the matrix pipe.
__global__ __launch_bounds__(256) void vq_kernel(const float* __restrict__ z,
                                                 const float* __restrict__ embed,
                                                 const _Float16* __restrict__ cb,
                                                 float* __restrict__ out,
                                                 int* __restrict__ idx_ws,
                                                 int* __restrict__ hist,
                                                 float* __restrict__ loss_part) {
    __shared__ float z_lds[32 * ZPAD];
    __shared__ float s_part[256];
    __shared__ float s_scale[32];
    __shared__ float red_val[32][8];
    __shared__ int   red_col[32][8];
    __shared__ int   s_idx[32];

    const int tid = threadIdx.x;
    const int T   = blockIdx.x;                 // row-tile: rows n = T*32..T*32+31
    const int bb  = T >> 5;                     // batch
    const int hh  = T & 31;                     // h (tile spans full w row)
    const size_t zbase = (size_t)bb * (DIM * 1024) + hh * 32; // (b,k,h,w) layout

    const int wave = tid >> 5, lane = tid & 31;

    // --- stage 32x256 z tile into LDS, accumulate sum-of-squares per row
    // thread: row = lane (fixed), k = p*8 + wave -> 128B coalesced per wave per pass
    float ss = 0.f;
    #pragma unroll
    for (int p = 0; p < 32; ++p) {
        const int k = p * 8 + wave;
        const float v = z[zbase + (size_t)k * 1024 + lane];
        z_lds[lane * ZPAD + k] = v;
        ss += v * v;
    }
    s_part[tid] = ss;
    __syncthreads();
    if (tid < 32) {
        float s = 0.f;
        #pragma unroll
        for (int i = 0; i < 8; ++i) s += s_part[tid + 32 * i];
        s_scale[tid] = 1.0f / fmaxf(sqrtf(s), 1e-12f);
    }
    __syncthreads();

    // --- build two A fragments per wave (rows 0-15 and 16-31), contiguous k order (k = 16h + i)
    const int row0 = lane & 15;
    const int row1 = row0 + 16;
    const int o    = (lane < 16) ? 0 : 16;
    const float zs0 = s_scale[row0];
    const float zs1 = s_scale[row1];
    v16h a0[8], a1[8];
    #pragma unroll
    for (int c = 0; c < 8; ++c) {
        const int kb = c * 32;
        #pragma unroll
        for (int i = 0; i < 8; ++i) {
            a0[c][i] = (_Float16)(z_lds[row0 * ZPAD + kb + o + i] * zs0);
            a1[c][i] = (_Float16)(z_lds[row1 * ZPAD + kb + o + i] * zs1);
        }
        #pragma unroll
        for (int i = 8; i < 16; ++i) {
            a0[c][i] = (_Float16)(z_lds[row0 * ZPAD + kb + o + i] * zs0);
            a1[c][i] = (_Float16)(z_lds[row1 * ZPAD + kb + o + i] * zs1);
        }
    }

    // --- GEMM + fused argmax over 4096 codes (similarity matrix never materialized)
    const v16h* cbp = (const v16h*)cb;
    float best0[8], best1[8];
    int   bcol0[8], bcol1[8];
    #pragma unroll
    for (int i = 0; i < 8; ++i) {
        best0[i] = -3.0e38f; bcol0[i] = 0;
        best1[i] = -3.0e38f; bcol1[i] = 0;
    }

    for (int t = wave; t < 256; t += 8) {
        v8f acc0 = {}, acc1 = {};
        #pragma unroll
        for (int c = 0; c < 8; ++c) {
            const v16h bv = cbp[(size_t)(t * 8 + c) * 32 + lane];  // coalesced 32B/lane
            acc0 = __builtin_amdgcn_wmma_f32_16x16x32_f16(
                false, a0[c], false, bv, (short)0, acc0, false, false);
            acc1 = __builtin_amdgcn_wmma_f32_16x16x32_f16(
                false, a1[c], false, bv, (short)0, acc1, false, false);
        }
        const int col = t * 16 + (lane & 15);
        #pragma unroll
        for (int i = 0; i < 8; ++i) {
            const float v0 = acc0[i];   // row = i + (lane<16?0:8)
            if (v0 > best0[i]) { best0[i] = v0; bcol0[i] = col; }
            const float v1 = acc1[i];   // row = 16 + i + (lane<16?0:8)
            if (v1 > best1[i]) { best1[i] = v1; bcol1[i] = col; }
        }
    }
    // cross-lane argmax within each 16-lane half (tie-break: lowest col == first max)
    #pragma unroll
    for (int i = 0; i < 8; ++i) {
        #pragma unroll
        for (int m = 1; m < 16; m <<= 1) {
            const float ov0 = __shfl_xor(best0[i], m, 32);
            const int   oc0 = __shfl_xor(bcol0[i], m, 32);
            if (ov0 > best0[i] || (ov0 == best0[i] && oc0 < bcol0[i])) { best0[i] = ov0; bcol0[i] = oc0; }
            const float ov1 = __shfl_xor(best1[i], m, 32);
            const int   oc1 = __shfl_xor(bcol1[i], m, 32);
            if (ov1 > best1[i] || (ov1 == best1[i] && oc1 < bcol1[i])) { best1[i] = ov1; bcol1[i] = oc1; }
        }
        if ((lane & 15) == 0) {
            const int rr = i + ((lane >> 4) << 3);
            red_val[rr][wave]      = best0[i];
            red_col[rr][wave]      = bcol0[i];
            red_val[rr + 16][wave] = best1[i];
            red_col[rr + 16][wave] = bcol1[i];
        }
    }
    __syncthreads();
    if (tid < 32) {
        float bv = -3.0e38f; int bc = 0;
        #pragma unroll
        for (int wv = 0; wv < 8; ++wv) {
            const float v = red_val[tid][wv]; const int c = red_col[tid][wv];
            if (v > bv || (v == bv && c < bc)) { bv = v; bc = c; }
        }
        s_idx[tid] = bc;
        const int n = T * 32 + tid;
        idx_ws[n] = bc;
        out[OUT_IDX + (size_t)n] = (float)bc;
        atomicAdd(&hist[bc], 1);
    }
    __syncthreads();

    // --- z_q gather, transposed store, deterministic loss partial
    float ls = 0.f;
    #pragma unroll
    for (int p = 0; p < 32; ++p) {
        const int k  = p * 8 + wave;
        const int id = s_idx[lane];
        const float e  = embed[(size_t)id * DIM + k];
        const float zv = z_lds[lane * ZPAD + k];
        out[OUT_ZQ + zbase + (size_t)k * 1024 + lane] = e;
        const float d = e - zv;
        ls += d * d;
    }
    s_part[tid] = ls;
    __syncthreads();
    for (int s = 128; s > 0; s >>= 1) { if (tid < s) s_part[tid] += s_part[tid + s]; __syncthreads(); }
    if (tid == 0) loss_part[T] = s_part[0];
}

// One-hot encodings: 268 MB of streaming stores (the HBM floor of this op).
// Row byte offset in d_out is only 8-aligned, so float2 stores.
__global__ __launch_bounds__(256) void enc_kernel(const int* __restrict__ idx,
                                                  float* __restrict__ enc) {
    const int n  = blockIdx.x;
    const int id = idx[n];
    float2* rowp = (float2*)(enc + (size_t)n * K_CODES);
    const int tid = threadIdx.x;
    #pragma unroll
    for (int j = 0; j < 8; ++j) {
        const int e2 = tid + j * 256;
        const int c0 = e2 << 1;
        float2 v;
        v.x = (c0     == id) ? 1.f : 0.f;
        v.y = (c0 + 1 == id) ? 1.f : 0.f;
        rowp[e2] = v;
    }
}

__global__ __launch_bounds__(256) void fin_kernel(const int* __restrict__ hist,
                                                  const float* __restrict__ loss_part,
                                                  float* __restrict__ out) {
    __shared__ float red[256];
    const int tid = threadIdx.x;
    float le = 0.f;
    for (int i = tid; i < 512; i += 256) le += loss_part[i];
    red[tid] = le;
    __syncthreads();
    for (int s = 128; s > 0; s >>= 1) { if (tid < s) red[tid] += red[tid + s]; __syncthreads(); }
    const float loss = red[0] * (1.25f / ((float)N_TOT * (float)DIM));
    __syncthreads();
    float pe = 0.f;
    for (int k = tid; k < K_CODES; k += 256) {
        const float p = (float)hist[k] * (1.0f / (float)N_TOT);
        pe += p * logf(p + 1e-10f);
    }
    red[tid] = pe;
    __syncthreads();
    for (int s = 128; s > 0; s >>= 1) { if (tid < s) red[tid] += red[tid + s]; __syncthreads(); }
    if (tid == 0) {
        out[OUT_LOSS] = loss;
        out[OUT_PERP] = expf(-red[0]);
    }
}

extern "C" void kernel_launch(void* const* d_in, const int* in_sizes, int n_in,
                              void* d_out, int out_size, void* d_ws, size_t ws_size,
                              hipStream_t stream) {
    const float* z     = (const float*)d_in[0];   // (16,256,32,32) fp32
    const float* embed = (const float*)d_in[1];   // (4096,256) fp32
    float* out = (float*)d_out;
    char*  ws  = (char*)d_ws;

    _Float16* cb        = (_Float16*)(ws + WS_CB);
    int*      idx       = (int*)(ws + WS_IDX);
    int*      hist      = (int*)(ws + WS_HIST);
    float*    loss_part = (float*)(ws + WS_LOSS);

    init_kernel<<<1, 256, 0, stream>>>(hist);
    cb_norm_kernel<<<K_CODES, 256, 0, stream>>>(embed, cb);
    vq_kernel<<<N_TOT / 32, 256, 0, stream>>>(z, embed, cb, out, idx, hist, loss_part);
    enc_kernel<<<N_TOT, 256, 0, stream>>>(idx, out + OUT_ENC);
    fin_kernel<<<1, 256, 0, stream>>>(hist, loss_part, out);
}
